// flux_GNN_53463752901238
// MI455X (gfx1250) — compile-verified, matrix operand order checked by reading, not run on hardware
//
#include <hip/hip_runtime.h>
#include <hip/hip_bf16.h>

typedef __attribute__((ext_vector_type(16))) __bf16 v16bf;
typedef __attribute__((ext_vector_type(8)))  float  v8f;

static constexpr int  DD    = 128;
static constexpr int  NB    = 2;
static constexpr int  NNODE = 50000;
static constexpr int  NCELL = 100000;
static constexpr int  NEDGE = 150000;

// fast reciprocal (v_rcp_f32) -- fine after bf16 WMMA precision
__device__ inline float fast_rcp(float x) { return __builtin_amdgcn_rcpf(x); }

// SiLU with native rcp: x * rcp(1 + exp(-x))
__device__ inline float fast_silu(float x) {
    return x * fast_rcp(1.f + __expf(-x));
}

// ---------------------------------------------------------------------------
// Fragment load: 16 bf16 values per lane, matching the CDNA5 16-bit A/B
// layout (lanes 0-15 hold K=kb..kb+7 in v0-3 and kb+16..kb+23 in v4-7;
// lanes 16-31 are offset by +8 in K).  Caller folds the +8/kstep into p.
// ---------------------------------------------------------------------------
__device__ inline v16bf frag_ld16(const __bf16* p) {
    union { v16bf v; uint4 q[2]; } u;
    u.q[0] = *(const uint4*)(p);
    u.q[1] = *(const uint4*)(p + 16);
    return u.v;
}

// ---------------------------------------------------------------------------
// Fused 2-layer MLP on a 16-row block staged in LDS.
//   hidden = silu(A[16xK1] @ W1t^T + b1)   (bf16 in LDS)
//   out    = hidden @ W2t^T + b2           (fp32 and/or bf16 to global)
// 8 waves, wave w owns output columns [16w, 16w+16).
// ---------------------------------------------------------------------------
template<int K1>
__device__ inline void mlp_tile(const __bf16* ldsA, __bf16* ldsH,
                                const __bf16* __restrict__ W1t,
                                const float*  __restrict__ b1,
                                const __bf16* __restrict__ W2t,
                                const float*  __restrict__ b2,
                                float* outF, __bf16* outB, long rowbase)
{
    const int tid     = threadIdx.x;
    const int wave    = tid >> 5;
    const int lane    = tid & 31;
    const int mcol    = lane & 15;          // row (A) / column (B,C)
    const int khalf   = (lane >> 4) << 3;   // 0 or 8
    const int rowoff  = (lane >> 4) << 3;   // C rows 0-7 vs 8-15
    const int colbase = wave * 16;
    const int col     = colbase + mcol;

    // ---- GEMM1 ----
    v8f acc = {0.f,0.f,0.f,0.f,0.f,0.f,0.f,0.f};
#pragma unroll
    for (int ks = 0; ks < K1 / 32; ++ks) {
        v16bf a = frag_ld16(ldsA + mcol * K1 + ks * 32 + khalf);
        v16bf b = frag_ld16(W1t + (long)col * K1 + ks * 32 + khalf);
        acc = __builtin_amdgcn_wmma_f32_16x16x32_bf16(
                  false, a, false, b, (short)0, acc, false, false);
    }
    {
        const float bias = b1[col];
#pragma unroll
        for (int r = 0; r < 8; ++r) {
            float x = acc[r] + bias;
            ldsH[(r + rowoff) * DD + col] = (__bf16)fast_silu(x);
        }
    }
    __syncthreads();

    // ---- GEMM2 (K = 128) ----
    v8f acc2 = {0.f,0.f,0.f,0.f,0.f,0.f,0.f,0.f};
#pragma unroll
    for (int ks = 0; ks < DD / 32; ++ks) {
        v16bf a = frag_ld16(ldsH + mcol * DD + ks * 32 + khalf);
        v16bf b = frag_ld16(W2t + (long)col * DD + ks * 32 + khalf);
        acc2 = __builtin_amdgcn_wmma_f32_16x16x32_bf16(
                   false, a, false, b, (short)0, acc2, false, false);
    }
    {
        const float bias = b2[col];
#pragma unroll
        for (int r = 0; r < 8; ++r) {
            float y = acc2[r] + bias;
            long row = rowbase + r + rowoff;
            if (outF) outF[row * DD + col] = y;
            if (outB) outB[row * DD + col] = (__bf16)y;
        }
    }
}

// ---------------------------------------------------------------------------
// Stage 1: cell MLP.  A row = mean of 3 gathered V rows.
// ---------------------------------------------------------------------------
__global__ void cell_mlp_kernel(const float* __restrict__ V,
                                const int*   __restrict__ cells,
                                const __bf16* W1t, const float* b1,
                                const __bf16* W2t, const float* b2,
                                __bf16* cellfeat)
{
    __shared__ __align__(16) __bf16 ldsA[16 * DD];
    __shared__ __align__(16) __bf16 ldsH[16 * DD];
    const long rowbase = (long)blockIdx.x * 16;

    for (int idx = threadIdx.x; idx < 16 * DD; idx += blockDim.x) {
        int  m = idx >> 7;
        int  k = idx & (DD - 1);
        long r = rowbase + m;
        int  b = (int)(r / NCELL);
        const int*   c  = cells + r * 3;
        const float* Vb = V + (long)b * NNODE * DD;
        float s = Vb[(long)c[0] * DD + k]
                + Vb[(long)c[1] * DD + k]
                + Vb[(long)c[2] * DD + k];
        ldsA[idx] = (__bf16)(s * (1.f / 3.f));
    }
    __syncthreads();
    mlp_tile<DD>(ldsA, ldsH, W1t, b1, W2t, b2, nullptr, cellfeat, rowbase);
}

// ---------------------------------------------------------------------------
// Stage 2: edge MLP.  A row = concat(E, cellfeat[left], cellfeat[right]).
// ---------------------------------------------------------------------------
__global__ void edge_mlp_kernel(const float* __restrict__ E,
                                const int*   __restrict__ e2c,
                                const __bf16* __restrict__ cellfeat,
                                const __bf16* W1t, const float* b1,
                                const __bf16* W2t, const float* b2,
                                float* edge_out)
{
    __shared__ __align__(16) __bf16 ldsA[16 * 3 * DD];
    __shared__ __align__(16) __bf16 ldsH[16 * DD];
    const long rowbase = (long)blockIdx.x * 16;

    for (int idx = threadIdx.x; idx < 16 * 3 * DD; idx += blockDim.x) {
        int  m = idx / (3 * DD);
        int  k = idx % (3 * DD);
        long r = rowbase + m;
        __bf16 val;
        if (k < DD) {
            val = (__bf16)E[r * DD + k];
        } else {
            int b  = (int)(r / NEDGE);
            int li = e2c[r * 2 + 0];
            int ri = e2c[r * 2 + 1];
            int l2 = (li >= 0) ? li : ri;
            int r2 = (ri >= 0) ? ri : li;
            int ci = (k < 2 * DD) ? l2 : r2;
            int kk = (k < 2 * DD) ? (k - DD) : (k - 2 * DD);
            val = cellfeat[((long)b * NCELL + ci) * DD + kk];
        }
        ldsA[idx] = val;
    }
    __syncthreads();
    mlp_tile<3 * DD>(ldsA, ldsH, W1t, b1, W2t, b2, edge_out, nullptr, rowbase);
}

// ---------------------------------------------------------------------------
// Scatter-mean accumulation: fp32 atomics into sum0/sum1 + counts.
// One thread per (edge, half-dim).
// ---------------------------------------------------------------------------
__global__ void scatter_kernel(const float* __restrict__ edge_out,
                               const int*   __restrict__ edges,
                               float* sum0, float* sum1,
                               float* cnt0, float* cnt1)
{
    long g = (long)blockIdx.x * blockDim.x + threadIdx.x;
    const long total = (long)NB * NEDGE * 64;
    if (g >= total) return;
    long e = g >> 6;
    int  t = (int)(g & 63);
    int  b = (int)(e / NEDGE);
    int  i0 = edges[e * 2 + 0];
    int  i1 = edges[e * 2 + 1];
    const float* er = edge_out + e * (long)DD;
    atomicAdd(&sum0[((long)b * NNODE + i0) * 64 + t], er[t]);
    atomicAdd(&sum1[((long)b * NNODE + i1) * 64 + t], er[64 + t]);
    if (t == 0) {
        atomicAdd(&cnt0[(long)b * NNODE + i0], 1.f);
        atomicAdd(&cnt1[(long)b * NNODE + i1], 1.f);
    }
}

// ---------------------------------------------------------------------------
// Stage 3: node MLP.  A row = concat(V, sum0/cnt0, sum1/cnt1).
// ---------------------------------------------------------------------------
__global__ void node_mlp_kernel(const float* __restrict__ V,
                                const float* __restrict__ sum0,
                                const float* __restrict__ sum1,
                                const float* __restrict__ cnt0,
                                const float* __restrict__ cnt1,
                                const __bf16* W1t, const float* b1,
                                const __bf16* W2t, const float* b2,
                                float* node_out)
{
    __shared__ __align__(16) __bf16 ldsA[16 * 2 * DD];
    __shared__ __align__(16) __bf16 ldsH[16 * DD];
    const long rowbase = (long)blockIdx.x * 16;

    for (int idx = threadIdx.x; idx < 16 * 2 * DD; idx += blockDim.x) {
        int  m = idx >> 8;
        int  k = idx & (2 * DD - 1);
        long r = rowbase + m;
        float v;
        if (k < DD) {
            v = V[r * DD + k];
        } else if (k < DD + 64) {
            v = sum0[r * 64 + (k - DD)] * fast_rcp(fmaxf(cnt0[r], 1.f));
        } else {
            v = sum1[r * 64 + (k - DD - 64)] * fast_rcp(fmaxf(cnt1[r], 1.f));
        }
        ldsA[idx] = (__bf16)v;
    }
    __syncthreads();
    mlp_tile<2 * DD>(ldsA, ldsH, W1t, b1, W2t, b2, node_out, nullptr, rowbase);
}

// ---------------------------------------------------------------------------
// Weight transpose + fp32 -> bf16:  Wt[n][k] = W[k][n]
// ---------------------------------------------------------------------------
__global__ void wtrans_kernel(const float* __restrict__ W, __bf16* Wt,
                              int K, int N)
{
    int idx = blockIdx.x * blockDim.x + threadIdx.x;
    if (idx >= K * N) return;
    int n = idx / K;
    int k = idx % K;
    Wt[idx] = (__bf16)W[k * N + n];
}

__global__ void zero_kernel(float* p, long n)
{
    long g = (long)blockIdx.x * blockDim.x + threadIdx.x;
    for (; g < n; g += (long)gridDim.x * blockDim.x) p[g] = 0.f;
}

// ---------------------------------------------------------------------------
extern "C" void kernel_launch(void* const* d_in, const int* in_sizes, int n_in,
                              void* d_out, int out_size, void* d_ws, size_t ws_size,
                              hipStream_t stream)
{
    const float* V     = (const float*)d_in[0];
    const float* E     = (const float*)d_in[1];
    const int*   edges = (const int*)d_in[2];
    const int*   cells = (const int*)d_in[3];
    const int*   e2c   = (const int*)d_in[4];
    const float* ceW1  = (const float*)d_in[5];
    const float* ceb1  = (const float*)d_in[6];
    const float* ceW2  = (const float*)d_in[7];
    const float* ceb2  = (const float*)d_in[8];
    const float* euW1  = (const float*)d_in[9];
    const float* eub1  = (const float*)d_in[10];
    const float* euW2  = (const float*)d_in[11];
    const float* eub2  = (const float*)d_in[12];
    const float* nuW1  = (const float*)d_in[13];
    const float* nub1  = (const float*)d_in[14];
    const float* nuW2  = (const float*)d_in[15];
    const float* nub2  = (const float*)d_in[16];

    float* node_out = (float*)d_out;                               // (B,N,D)
    float* edge_out = node_out + (long)NB * NNODE * DD;            // (B,Ne,D)

    // workspace carve-out
    char* ws = (char*)d_ws;
    size_t off = 0;
    auto carve = [&](size_t bytes) -> void* {
        void* p = ws + off;
        off += (bytes + 255) & ~(size_t)255;
        return p;
    };
    __bf16* ceW1t    = (__bf16*)carve((size_t)DD * DD * 2);
    __bf16* ceW2t    = (__bf16*)carve((size_t)DD * DD * 2);
    __bf16* euW1t    = (__bf16*)carve((size_t)DD * 3 * DD * 2);
    __bf16* euW2t    = (__bf16*)carve((size_t)DD * DD * 2);
    __bf16* nuW1t    = (__bf16*)carve((size_t)DD * 2 * DD * 2);
    __bf16* nuW2t    = (__bf16*)carve((size_t)DD * DD * 2);
    __bf16* cellfeat = (__bf16*)carve((size_t)NB * NCELL * DD * 2);
    float*  sum0     = (float*)carve((size_t)NB * NNODE * 64 * 4);
    float*  sum1     = (float*)carve((size_t)NB * NNODE * 64 * 4);
    float*  cnt0     = (float*)carve((size_t)NB * NNODE * 4);
    float*  cnt1     = (float*)carve((size_t)NB * NNODE * 4);
    (void)ws_size; (void)in_sizes; (void)n_in; (void)out_size;

    // 0) weight transposes (bf16)
    wtrans_kernel<<<(DD * DD + 255) / 256, 256, 0, stream>>>(ceW1, ceW1t, DD, DD);
    wtrans_kernel<<<(DD * DD + 255) / 256, 256, 0, stream>>>(ceW2, ceW2t, DD, DD);
    wtrans_kernel<<<(3 * DD * DD + 255) / 256, 256, 0, stream>>>(euW1, euW1t, 3 * DD, DD);
    wtrans_kernel<<<(DD * DD + 255) / 256, 256, 0, stream>>>(euW2, euW2t, DD, DD);
    wtrans_kernel<<<(2 * DD * DD + 255) / 256, 256, 0, stream>>>(nuW1, nuW1t, 2 * DD, DD);
    wtrans_kernel<<<(DD * DD + 255) / 256, 256, 0, stream>>>(nuW2, nuW2t, DD, DD);

    // 1) zero scatter accumulators (sum0,sum1,cnt0,cnt1 are contiguous)
    {
        long nz = (long)NB * NNODE * 64 * 2 + (long)NB * NNODE * 2
                  + 128; // padding slack from 256B rounding
        zero_kernel<<<4096, 256, 0, stream>>>(sum0, nz);
    }

    // 2) cell MLP -> cellfeat (bf16)
    cell_mlp_kernel<<<(NB * NCELL) / 16, 256, 0, stream>>>(
        V, cells, ceW1t, ceb1, ceW2t, ceb2, cellfeat);

    // 3) edge MLP -> edge_out (fp32, second output section)
    edge_mlp_kernel<<<(NB * NEDGE) / 16, 256, 0, stream>>>(
        E, e2c, cellfeat, euW1t, eub1, euW2t, eub2, edge_out);

    // 4) scatter-mean accumulation
    {
        long total = (long)NB * NEDGE * 64;
        scatter_kernel<<<(int)((total + 255) / 256), 256, 0, stream>>>(
            edge_out, edges, sum0, sum1, cnt0, cnt1);
    }

    // 5) node MLP -> node_out (fp32, first output section)
    node_mlp_kernel<<<(NB * NNODE) / 16, 256, 0, stream>>>(
        V, sum0, sum1, cnt0, cnt1, nuW1t, nub1, nuW2t, nub2, node_out);
}
